// MultiHeadAttention2_81097572483411
// MI455X (gfx1250) — matrix operand order repaired and checked by reading, not compiled, not run on hardware
//
#include <hip/hip_runtime.h>

#define DEV __device__ __forceinline__

typedef __attribute__((ext_vector_type(16))) _Float16 v16h;
typedef __attribute__((ext_vector_type(8)))  _Float16 v8h;
typedef __attribute__((ext_vector_type(8)))  float    v8f;
typedef int b128_t __attribute__((vector_size(16)));

constexpr int B_ = 8, S_ = 1024, D_ = 1024, H_ = 16, DEPTH_ = 64;
constexpr int M_ = B_ * S_;              // 8192
constexpr float SCALE_ = 0.125f;         // 1/sqrt(64)

// ---------------------------------------------------------------------------
// CDNA5 async memory->LDS copy (ASYNCcnt path), 16 bytes per lane.
// Builtin signature (from probe diagnostic): param0 = AS(1) b128*, so:
//   (global b128*, shared b128*, imm offset, imm cpol)
// ---------------------------------------------------------------------------
#if __has_builtin(__builtin_amdgcn_global_load_async_to_lds_b128)
#define HAVE_ASYNC_BUILTIN 1
#endif
#if __has_builtin(__builtin_amdgcn_s_wait_asynccnt)
#define HAVE_ASYNC_WAIT_BUILTIN 1
#endif

DEV void async_cp16(void* lds, const void* g) {
#ifdef HAVE_ASYNC_BUILTIN
  __builtin_amdgcn_global_load_async_to_lds_b128(
      (__attribute__((address_space(1))) b128_t*)g,
      (__attribute__((address_space(3))) b128_t*)lds, 0, 0);
#else
  unsigned l = (unsigned)(size_t)lds;
  asm volatile("global_load_async_to_lds_b128 %0, %1, off"
               :: "v"(l), "v"(g) : "memory");
#endif
}

DEV void async_wait0() {
#ifdef HAVE_ASYNC_WAIT_BUILTIN
  __builtin_amdgcn_s_wait_asynccnt(0);
#else
  asm volatile("s_wait_asynccnt 0x0" ::: "memory");
#endif
}

// ---------------------------------------------------------------------------
// WMMA helpers — layouts per CDNA5 ISA 7.12.2 (wave32)
// ---------------------------------------------------------------------------
DEV v16h frag_a(const _Float16* tile, int lda, int row0, int k0, int lane) {
  int row = row0 + (lane & 15);
  int ko  = k0 + ((lane >> 4) << 3);
  const _Float16* p = tile + row * lda + ko;
  v8h lo = *(const v8h*)(p);
  v8h hi = *(const v8h*)(p + 16);
  v16h a;
#pragma unroll
  for (int i = 0; i < 8; ++i) { a[i] = lo[i]; a[i + 8] = hi[i]; }
  return a;
}

DEV v16h frag_b(const _Float16* tileT, int ldb, int col0, int k0, int lane) {
  int col = col0 + (lane & 15);
  int ko  = k0 + ((lane >> 4) << 3);
  const _Float16* p = tileT + col * ldb + ko;
  v8h lo = *(const v8h*)(p);
  v8h hi = *(const v8h*)(p + 16);
  v16h b;
#pragma unroll
  for (int i = 0; i < 8; ++i) { b[i] = lo[i]; b[i + 8] = hi[i]; }
  return b;
}

DEV v8f wmma32(v16h a, v16h b, v8f c) {
  return __builtin_amdgcn_wmma_f32_16x16x32_f16(false, a, false, b,
                                                (short)0, c, false, false);
}

// ---------------------------------------------------------------------------
// Projection GEMM: Y = X(f32)[8192x1024] @ W(f32)[1024x1024], output f16 in
// head-split layout Yh[(b*H+h)*S + s][64]. f32->f16 conversion fused into
// LDS staging (VALU path required for the convert).
// ---------------------------------------------------------------------------
__global__ __launch_bounds__(256)
void proj_gemm_kernel(const float* __restrict__ X, const float* __restrict__ W,
                      _Float16* __restrict__ Yh) {
  constexpr int BM = 128, BN = 128, BK = 64, LDA = BK + 8, LDB = BK + 8;
  __shared__ __align__(16) _Float16 At[BM * LDA];
  __shared__ __align__(16) _Float16 Bt[BN * LDB];   // [col][k]

  const int m0 = blockIdx.x * BM, n0 = blockIdx.y * BN;
  const int tid = threadIdx.x, lane = tid & 31;
  const int w = tid >> 5, wm = w >> 2, wn = w & 3;

  v8f acc[4][2];
#pragma unroll
  for (int i = 0; i < 4; ++i)
#pragma unroll
    for (int j = 0; j < 2; ++j) acc[i][j] = (v8f)(0.0f);

  for (int k0 = 0; k0 < D_; k0 += BK) {
    {   // stage A: 128x64 f32 -> f16, 32 elems/thread
      int r = tid >> 1, c = (tid & 1) * 32;
      const float4* src = (const float4*)(X + (size_t)(m0 + r) * D_ + k0 + c);
      _Float16* dst = At + r * LDA + c;
#pragma unroll
      for (int i = 0; i < 8; ++i) {
        float4 f = src[i];
        dst[i * 4 + 0] = (_Float16)f.x; dst[i * 4 + 1] = (_Float16)f.y;
        dst[i * 4 + 2] = (_Float16)f.z; dst[i * 4 + 3] = (_Float16)f.w;
      }
      if (k0 + BK < D_)   // prefetch next A tile (global_prefetch_b8)
        __builtin_prefetch(X + (size_t)(m0 + r) * D_ + k0 + BK + c, 0, 3);
    }
    {   // stage B transposed: W rows k0..k0+64, cols n0..n0+128
      int kr = tid >> 2, c = (tid & 3) * 32;
      const float4* src = (const float4*)(W + (size_t)(k0 + kr) * D_ + n0 + c);
#pragma unroll
      for (int i = 0; i < 8; ++i) {
        float4 f = src[i];
        Bt[(c + i * 4 + 0) * LDB + kr] = (_Float16)f.x;
        Bt[(c + i * 4 + 1) * LDB + kr] = (_Float16)f.y;
        Bt[(c + i * 4 + 2) * LDB + kr] = (_Float16)f.z;
        Bt[(c + i * 4 + 3) * LDB + kr] = (_Float16)f.w;
      }
      if (k0 + BK < D_)
        __builtin_prefetch(W + (size_t)(k0 + BK + kr) * D_ + n0 + c, 0, 3);
    }
    __syncthreads();
#pragma unroll
    for (int ks = 0; ks < BK; ks += 32) {
      v16h bf0 = frag_b(Bt, LDB, wn * 32,      ks, lane);
      v16h bf1 = frag_b(Bt, LDB, wn * 32 + 16, ks, lane);
#pragma unroll
      for (int i = 0; i < 4; ++i) {
        v16h af = frag_a(At, LDA, wm * 64 + i * 16, ks, lane);
        acc[i][0] = wmma32(af, bf0, acc[i][0]);
        acc[i][1] = wmma32(af, bf1, acc[i][1]);
      }
    }
    __syncthreads();
  }
  const int hi8 = (lane >> 4) << 3;
#pragma unroll
  for (int i = 0; i < 4; ++i)
#pragma unroll
    for (int j = 0; j < 2; ++j) {
      int n = n0 + wn * 32 + j * 16 + (lane & 15);
      int h = n >> 6, d = n & 63;
#pragma unroll
      for (int r = 0; r < 8; ++r) {
        int m = m0 + wm * 64 + i * 16 + r + hi8;
        int b = m >> 10, s = m & 1023;
        Yh[(((size_t)(b * H_ + h)) * S_ + s) * DEPTH_ + d] = (_Float16)acc[i][j][r];
      }
    }
}

// ---------------------------------------------------------------------------
// Output GEMM: out = ctx(f16)[8192x1024] @ Wo(f32)[1024x1024] + bo, f32 out.
// A tile is a pure f16 copy -> async memory->LDS path.
// ---------------------------------------------------------------------------
__global__ __launch_bounds__(256)
void out_gemm_kernel(const _Float16* __restrict__ Xh, const float* __restrict__ W,
                     const float* __restrict__ bo, float* __restrict__ out) {
  constexpr int BM = 128, BN = 128, BK = 64, LDA = BK + 8, LDB = BK + 8;
  __shared__ __align__(16) _Float16 At[BM * LDA];
  __shared__ __align__(16) _Float16 Bt[BN * LDB];

  const int m0 = blockIdx.x * BM, n0 = blockIdx.y * BN;
  const int tid = threadIdx.x, lane = tid & 31;
  const int w = tid >> 5, wm = w >> 2, wn = w & 3;

  v8f acc[4][2];
#pragma unroll
  for (int i = 0; i < 4; ++i)
#pragma unroll
    for (int j = 0; j < 2; ++j) acc[i][j] = (v8f)(0.0f);

  for (int k0 = 0; k0 < D_; k0 += BK) {
    {   // stage A (f16 copy): async global->LDS, 4x b128 per thread
      int r = tid >> 1, c = (tid & 1) * 32;
      const _Float16* src = Xh + (size_t)(m0 + r) * D_ + k0 + c;
      _Float16* dst = At + r * LDA + c;
#pragma unroll
      for (int i = 0; i < 4; ++i) async_cp16(dst + i * 8, src + i * 8);
    }
    {   // stage B transposed, f32 -> f16 (VALU convert path)
      int kr = tid >> 2, c = (tid & 3) * 32;
      const float4* src = (const float4*)(W + (size_t)(k0 + kr) * D_ + n0 + c);
#pragma unroll
      for (int i = 0; i < 8; ++i) {
        float4 f = src[i];
        Bt[(c + i * 4 + 0) * LDB + kr] = (_Float16)f.x;
        Bt[(c + i * 4 + 1) * LDB + kr] = (_Float16)f.y;
        Bt[(c + i * 4 + 2) * LDB + kr] = (_Float16)f.z;
        Bt[(c + i * 4 + 3) * LDB + kr] = (_Float16)f.w;
      }
    }
    async_wait0();
    __syncthreads();
#pragma unroll
    for (int ks = 0; ks < BK; ks += 32) {
      v16h bf0 = frag_b(Bt, LDB, wn * 32,      ks, lane);
      v16h bf1 = frag_b(Bt, LDB, wn * 32 + 16, ks, lane);
#pragma unroll
      for (int i = 0; i < 4; ++i) {
        v16h af = frag_a(At, LDA, wm * 64 + i * 16, ks, lane);
        acc[i][0] = wmma32(af, bf0, acc[i][0]);
        acc[i][1] = wmma32(af, bf1, acc[i][1]);
      }
    }
    __syncthreads();
  }
  const int hi8 = (lane >> 4) << 3;
#pragma unroll
  for (int i = 0; i < 4; ++i)
#pragma unroll
    for (int j = 0; j < 2; ++j) {
      int n = n0 + wn * 32 + j * 16 + (lane & 15);
      float bias = bo[n];
#pragma unroll
      for (int r = 0; r < 8; ++r) {
        int m = m0 + wm * 64 + i * 16 + r + hi8;
        out[(size_t)m * D_ + n] = acc[i][j][r] + bias;
      }
    }
}

// ---------------------------------------------------------------------------
// Pass A: per (b,h) x 128-query block, compute QK^T tiles with WMMA and
// reduce online (max, sum-exp) partials. Q/K tiles staged via async->LDS.
// ---------------------------------------------------------------------------
__global__ __launch_bounds__(256)
void attn_stats_kernel(const _Float16* __restrict__ Qh,
                       const _Float16* __restrict__ Kh,
                       float* __restrict__ partials) {
  constexpr int LD = DEPTH_ + 8;   // 72
  __shared__ __align__(16) _Float16 Qt[128 * LD];
  __shared__ __align__(16) _Float16 Kt[64 * LD];
  __shared__ float redm[8], reds[8];

  const int bh = blockIdx.x >> 3, qb = blockIdx.x & 7;
  const int tid = threadIdx.x, lane = tid & 31;
  const int w = tid >> 5, wm = w >> 2, wn = w & 3;
  const _Float16* Qbase = Qh + ((size_t)bh * S_ + qb * 128) * DEPTH_;

  {   // stage Q once: async f16 copy
    int r = tid >> 1, c = (tid & 1) * 32;
    const _Float16* src = Qbase + r * DEPTH_ + c;
    _Float16* dst = Qt + r * LD + c;
#pragma unroll
    for (int i = 0; i < 4; ++i) async_cp16(dst + i * 8, src + i * 8);
  }

  float m = -3.0e38f, ssum = 0.0f;
  for (int kt = 0; kt < S_ / 64; ++kt) {
    __syncthreads();
    {   // stage K tile: async f16 copy (rows already B^T layout [key][depth])
      int r = tid >> 2, c = (tid & 3) * 16;
      const _Float16* src = Kh + ((size_t)bh * S_ + kt * 64 + r) * DEPTH_ + c;
      _Float16* dst = Kt + r * LD + c;
      async_cp16(dst, src);
      async_cp16(dst + 8, src + 8);
      if (kt + 1 < S_ / 64)
        __builtin_prefetch(src + 64 * DEPTH_, 0, 3);
    }
    async_wait0();
    __syncthreads();

    v8f sacc[4];
#pragma unroll
    for (int i = 0; i < 4; ++i) sacc[i] = (v8f)(0.0f);
#pragma unroll
    for (int ks = 0; ks < DEPTH_; ks += 32) {
      v16h bf = frag_b(Kt, LD, wn * 16, ks, lane);
#pragma unroll
      for (int i = 0; i < 4; ++i) {
        v16h af = frag_a(Qt, LD, wm * 64 + i * 16, ks, lane);
        sacc[i] = wmma32(af, bf, sacc[i]);
      }
    }
    // online (max, sum) over this tile's 32 values per lane
    float lm = -3.0e38f;
#pragma unroll
    for (int i = 0; i < 4; ++i)
#pragma unroll
      for (int r = 0; r < 8; ++r) {
        sacc[i][r] *= SCALE_;
        lm = fmaxf(lm, sacc[i][r]);
      }
    float nM = fmaxf(m, lm), ls = 0.0f;
#pragma unroll
    for (int i = 0; i < 4; ++i)
#pragma unroll
      for (int r = 0; r < 8; ++r) ls += __expf(sacc[i][r] - nM);
    ssum = ssum * __expf(m - nM) + ls;
    m = nM;
  }
  // wave32 reduce of (m, ssum)
#pragma unroll
  for (int off = 16; off > 0; off >>= 1) {
    float om = __shfl_xor(m, off, 32);
    float os = __shfl_xor(ssum, off, 32);
    float nM = fmaxf(m, om);
    ssum = ssum * __expf(m - nM) + os * __expf(om - nM);
    m = nM;
  }
  if (lane == 0) { redm[w] = m; reds[w] = ssum; }
  __syncthreads();
  if (tid == 0) {
    float M = -3.0e38f, Ssum = 0.0f;
    for (int i = 0; i < 8; ++i) {
      float nM = fmaxf(M, redm[i]);
      Ssum = Ssum * __expf(M - nM) + reds[i] * __expf(redm[i] - nM);
      M = nM;
    }
    partials[blockIdx.x * 2 + 0] = M;
    partials[blockIdx.x * 2 + 1] = Ssum;
  }
}

// Merge 8 query-block partials per (b,h) into global (max, sum).
__global__ void stats_merge_kernel(const float* __restrict__ partials,
                                   float* __restrict__ gstats) {
  int bh = threadIdx.x;
  if (bh >= B_ * H_) return;
  float M = -3.0e38f, Ssum = 0.0f;
  for (int i = 0; i < 8; ++i) {
    float pm = partials[(bh * 8 + i) * 2 + 0];
    float ps = partials[(bh * 8 + i) * 2 + 1];
    float nM = fmaxf(M, pm);
    Ssum = Ssum * __expf(M - nM) + ps * __expf(pm - nM);
    M = nM;
  }
  gstats[bh * 2 + 0] = M;
  gstats[bh * 2 + 1] = Ssum;
}

// ---------------------------------------------------------------------------
// Pass B: recompute QK^T tiles, P = exp(l - gmax); write attn = P/gsum to
// d_out; accumulate ctx += P @ V in registers; ctx/gsum -> f16 concat layout.
// ---------------------------------------------------------------------------
__global__ __launch_bounds__(256)
void attn_ctx_kernel(const _Float16* __restrict__ Qh,
                     const _Float16* __restrict__ Kh,
                     const _Float16* __restrict__ Vh,
                     const float* __restrict__ gstats,
                     _Float16* __restrict__ ctxc,
                     float* __restrict__ attn) {
  constexpr int LD = DEPTH_ + 8;   // 72
  __shared__ __align__(16) _Float16 Qt[128 * LD];
  __shared__ __align__(16) _Float16 Kt[64 * LD];
  __shared__ __align__(16) _Float16 Vt[64 * LD];    // [depth][key] transposed
  __shared__ __align__(16) _Float16 Pt[128 * LD];   // [qrow][key]

  const int bh = blockIdx.x >> 3, qb = blockIdx.x & 7;
  const int b = bh >> 4, h = bh & 15;
  const int tid = threadIdx.x, lane = tid & 31;
  const int w = tid >> 5, wm = w >> 2, wn = w & 3;
  const int hi8 = (lane >> 4) << 3;

  {   // stage Q once: async f16 copy
    int r = tid >> 1, c = (tid & 1) * 32;
    const _Float16* src = Qh + ((size_t)bh * S_ + qb * 128 + r) * DEPTH_ + c;
    _Float16* dst = Qt + r * LD + c;
#pragma unroll
    for (int i = 0; i < 4; ++i) async_cp16(dst + i * 8, src + i * 8);
  }
  const float gm = gstats[bh * 2 + 0];
  const float ginv = 1.0f / gstats[bh * 2 + 1];

  v8f cacc[4];
#pragma unroll
  for (int i = 0; i < 4; ++i) cacc[i] = (v8f)(0.0f);

  for (int kt = 0; kt < S_ / 64; ++kt) {
    __syncthreads();
    {   // stage K tile: async f16 copy
      int r = tid >> 2, c = (tid & 3) * 16;
      const _Float16* src = Kh + ((size_t)bh * S_ + kt * 64 + r) * DEPTH_ + c;
      _Float16* dst = Kt + r * LD + c;
      async_cp16(dst, src);
      async_cp16(dst + 8, src + 8);
      if (kt + 1 < S_ / 64)
        __builtin_prefetch(src + 64 * DEPTH_, 0, 3);
    }
    {   // stage V tile transposed (VALU path): Vt[depth][key]
      int key = tid >> 2, c = (tid & 3) * 16;
      const _Float16* src = Vh + ((size_t)bh * S_ + kt * 64 + key) * DEPTH_ + c;
#pragma unroll
      for (int cc = 0; cc < 2; ++cc) {
        v8h vv = *(const v8h*)(src + cc * 8);
#pragma unroll
        for (int i = 0; i < 8; ++i) Vt[(c + cc * 8 + i) * LD + key] = vv[i];
      }
      if (kt + 1 < S_ / 64)
        __builtin_prefetch(src + 64 * DEPTH_, 0, 3);
    }
    async_wait0();
    __syncthreads();

    // S tile = Q @ K^T (128x64)
    v8f sacc[4];
#pragma unroll
    for (int i = 0; i < 4; ++i) sacc[i] = (v8f)(0.0f);
#pragma unroll
    for (int ks = 0; ks < DEPTH_; ks += 32) {
      v16h bf = frag_b(Kt, LD, wn * 16, ks, lane);
#pragma unroll
      for (int i = 0; i < 4; ++i) {
        v16h af = frag_a(Qt, LD, wm * 64 + i * 16, ks, lane);
        sacc[i] = wmma32(af, bf, sacc[i]);
      }
    }
    // P = exp(l - gm): store f16 to Pt, fp32 attn/gsum to d_out
    const int col = wn * 16 + (lane & 15);
    const int keyg = kt * 64 + col;
#pragma unroll
    for (int i = 0; i < 4; ++i)
#pragma unroll
      for (int r = 0; r < 8; ++r) {
        float p = __expf(sacc[i][r] * SCALE_ - gm);
        int row = wm * 64 + i * 16 + r + hi8;
        Pt[row * LD + col] = (_Float16)p;
        attn[(((size_t)bh * S_) + qb * 128 + row) * S_ + keyg] = p * ginv;
      }
    __syncthreads();

    // ctx += P (128x64) @ V (64x64)
#pragma unroll
    for (int ks = 0; ks < 64; ks += 32) {
      v16h bf = frag_b(Vt, LD, wn * 16, ks, lane);
#pragma unroll
      for (int i = 0; i < 4; ++i) {
        v16h af = frag_a(Pt, LD, wm * 64 + i * 16, ks, lane);
        cacc[i] = wmma32(af, bf, cacc[i]);
      }
    }
  }
  // ctx / gsum -> f16, concat layout [b][s][h*64+d]
#pragma unroll
  for (int i = 0; i < 4; ++i) {
    int d = wn * 16 + (lane & 15);
#pragma unroll
    for (int r = 0; r < 8; ++r) {
      int srow = qb * 128 + wm * 64 + i * 16 + r + hi8;
      ctxc[((size_t)b * S_ + srow) * D_ + h * DEPTH_ + d] = (_Float16)(cacc[i][r] * ginv);
    }
  }
}

// ---------------------------------------------------------------------------
extern "C" void kernel_launch(void* const* d_in, const int* in_sizes, int n_in,
                              void* d_out, int out_size, void* d_ws, size_t ws_size,
                              hipStream_t stream) {
  const float* v  = (const float*)d_in[0];
  const float* k  = (const float*)d_in[1];
  const float* q  = (const float*)d_in[2];
  const float* Wq = (const float*)d_in[3];
  const float* Wk = (const float*)d_in[4];
  const float* Wv = (const float*)d_in[5];
  const float* Wo = (const float*)d_in[6];
  const float* bo = (const float*)d_in[7];

  float* out  = (float*)d_out;
  float* attn = out + (size_t)M_ * D_;     // tuple output #2: [B,H,S,S] fp32

  _Float16* Qh   = (_Float16*)d_ws;                  // head-split f16, 16 MB each
  _Float16* Kh   = Qh   + (size_t)M_ * D_;
  _Float16* Vh   = Kh   + (size_t)M_ * D_;
  _Float16* ctxc = Vh   + (size_t)M_ * D_;
  float* partials = (float*)(ctxc + (size_t)M_ * D_);   // [128][8][2]
  float* gstats   = partials + B_ * H_ * 8 * 2;         // [128][2]

  dim3 gg(M_ / 128, D_ / 128), blk(256);
  proj_gemm_kernel<<<gg, blk, 0, stream>>>(q, Wq, Qh);
  proj_gemm_kernel<<<gg, blk, 0, stream>>>(k, Wk, Kh);
  proj_gemm_kernel<<<gg, blk, 0, stream>>>(v, Wv, Vh);

  attn_stats_kernel<<<B_ * H_ * (S_ / 128), blk, 0, stream>>>(Qh, Kh, partials);
  stats_merge_kernel<<<1, B_ * H_, 0, stream>>>(partials, gstats);
  attn_ctx_kernel<<<B_ * H_ * (S_ / 128), blk, 0, stream>>>(Qh, Kh, Vh, gstats, ctxc, attn);

  out_gemm_kernel<<<gg, blk, 0, stream>>>(ctxc, Wo, bo, out);
}
